// AttentionTSSA_13486197310326
// MI455X (gfx1250) — compile-verified, matrix operand order checked by reading, not run on hardware
//
#include <hip/hip_runtime.h>

#define BSZ  16
#define NTOK 4096
#define DIM  768
#define NH   12
#define DH   64
#define MTOT (BSZ * NTOK)

typedef __attribute__((ext_vector_type(16))) __bf16 v16bf;
typedef __attribute__((ext_vector_type(8)))  float  v8f;
typedef __attribute__((ext_vector_type(4)))  unsigned int u32x4;
typedef __attribute__((ext_vector_type(8)))  int i32x8;
typedef __attribute__((ext_vector_type(4)))  int i32x4;

#if __has_builtin(__builtin_amdgcn_tensor_load_to_lds)
#define USE_TDM 1
#else
#define USE_TDM 0
#endif

__device__ __forceinline__ __bf16 f2bf(float f) {
    unsigned u = __builtin_bit_cast(unsigned, f);
    u = (u + 0x7FFFu + ((u >> 16) & 1u)) >> 16;
    unsigned short s = (unsigned short)u;
    return __builtin_bit_cast(__bf16, s);
}
__device__ __forceinline__ float bf2f(__bf16 h) {
    unsigned short s = __builtin_bit_cast(unsigned short, h);
    unsigned u = ((unsigned)s) << 16;
    return __builtin_bit_cast(float, u);
}

// 16-bit A-matrix 16x32 fragment (ISA 7.12.2): lane holds M=lane%16,
// K in [base, base+8) and [base+16, base+24), base = 8*(lane>=16).
__device__ __forceinline__ v16bf load_frag(const __bf16* rowp, int base) {
    union { uint4 q[2]; v16bf v; } u;
    u.q[0] = *(const uint4*)(rowp + base);
    u.q[1] = *(const uint4*)(rowp + base + 16);
    return u.v;
}

#if USE_TDM
// TDM 2D tile load: bf16 (data_size=2B) tile_rows x tile_cols from a row-major
// tensor (stride0 elements/row) into LDS at ldsoff. D# per ISA ch.8 (08_async_tensor).
__device__ __forceinline__ void tdm_load_2d(unsigned ldsoff, const __bf16* gptr,
                                            unsigned tile_rows, unsigned tile_cols,
                                            unsigned tdim0, unsigned tdim1,
                                            unsigned stride0) {
    unsigned long long ga = (unsigned long long)(size_t)gptr;
    u32x4 g0 = { 1u,                                   // count=1, user desc
                 ldsoff,                               // lds_addr
                 (unsigned)ga,                         // global_addr[31:0]
                 (unsigned)((ga >> 32) & 0x01FFFFFFu) | (2u << 30) };  // [56:32] | type=2
    i32x8 g1 = { (int)0x00010000u,                     // wg_mask=0, data_size=1 (2 bytes)
                 (int)((tdim0 & 0xFFFFu) << 16),       // tensor_dim0[15:0]
                 (int)(((tdim0 >> 16) & 0xFFFFu) | ((tdim1 & 0xFFFFu) << 16)),
                 (int)(((tdim1 >> 16) & 0xFFFFu) | ((tile_cols & 0xFFFFu) << 16)),
                 (int)(tile_rows & 0xFFFFu),           // tile_dim1; tile_dim2=0
                 (int)stride0,                         // tensor_dim0_stride[31:0]
                 0, 0 };
    i32x4 z4 = {0, 0, 0, 0};
#if __clang_major__ >= 23
    i32x8 z8 = {0, 0, 0, 0, 0, 0, 0, 0};
    __builtin_amdgcn_tensor_load_to_lds(g0, g1, z4, z4, z8, 0);
#else
    __builtin_amdgcn_tensor_load_to_lds(g0, g1, z4, z4, 0);
#endif
}
__device__ __forceinline__ unsigned lds_off(const void* p) {
    return (unsigned)(size_t)p;   // generic LDS pointer: addr[31:0] = LDS byte address
}
#endif

// C = A(65536xK) * W^T ; W pre-split to hi/lo bf16 (wh/wl, 768x768 row-major).
// mode==1: A element (m,c) scaled on the fly by -Pi[b,h(c),n]*attn[b,c], bias added.
__launch_bounds__(256)
__global__ void gemm_bf16x3_kernel(const float* __restrict__ A,
                                   const __bf16* __restrict__ wh,
                                   const __bf16* __restrict__ wl,
                                   float* __restrict__ C,
                                   int mode,
                                   const float* __restrict__ Pi,
                                   const float* __restrict__ attn,
                                   const float* __restrict__ bias) {
    __shared__ __align__(16) __bf16 Ah[128][32];
    __shared__ __align__(16) __bf16 Al[128][32];
#if USE_TDM
    __shared__ __align__(16) __bf16 Bh[2][64][32];   // double-buffered (TDM)
    __shared__ __align__(16) __bf16 Bl[2][64][32];
#else
    __shared__ __align__(16) __bf16 Bh[1][64][32];
    __shared__ __align__(16) __bf16 Bl[1][64][32];
#endif

    const int tid  = threadIdx.x;
    const int lane = tid & 31;
    const int wv   = tid >> 5;          // 8 waves
    const int wm   = wv >> 1;           // 4 wave-rows (32 rows each)
    const int wn   = wv & 1;            // 2 wave-cols (32 cols each)
    const int row0 = blockIdx.y * 128;
    const int col0 = blockIdx.x * 64;
    const int lr   = lane & 15;
    const int base = (lane >> 4) * 8;

    v8f acc[2][2] = {};
    int cur = 0;

#if USE_TDM
    // preload B tiles for kk=0 (one TDM descriptor per producer wave)
    if (wv == 0)
        tdm_load_2d(lds_off(&Bh[0][0][0]), wh + (size_t)col0 * DIM, 64, 32, DIM, DIM, DIM);
    else if (wv == 1)
        tdm_load_2d(lds_off(&Bl[0][0][0]), wl + (size_t)col0 * DIM, 64, 32, DIM, DIM, DIM);
#endif

    for (int kk = 0; kk < DIM; kk += 32) {
        __syncthreads();   // all waves done reading previous A and B[cur^1]
        // ---- stage A tile 128x32 fp32 -> hi/lo bf16 (1024 float4, 4/thread)
        for (int i = 0; i < 4; ++i) {
            int q  = tid + 256 * i;
            int r  = q >> 3;
            int kc = (q & 7) << 2;
            int gm = row0 + r;
            const float* ap = A + (size_t)gm * DIM + kk + kc;
            if (i == 0 && kk + 32 < DIM)
                __builtin_prefetch(ap + 32, 0, 3);    // global_prefetch_b8: next K-slice
            float4 v = *(const float4*)ap;
            float f[4] = {v.x, v.y, v.z, v.w};
            if (mode) {
                int b  = gm >> 12;
                int nt = gm & (NTOK - 1);
                int c  = kk + kc;
                float p = -Pi[(size_t)(b * NH + (c >> 6)) * NTOK + nt];
                float4 a4 = *(const float4*)(attn + b * DIM + c);
                f[0] *= p * a4.x; f[1] *= p * a4.y;
                f[2] *= p * a4.z; f[3] *= p * a4.w;
            }
            for (int j = 0; j < 4; ++j) {
                __bf16 h = f2bf(f[j]);
                Ah[r][kc + j] = h;
                Al[r][kc + j] = f2bf(f[j] - bf2f(h));
            }
        }
#if USE_TDM
        if (wv < 2) __builtin_amdgcn_s_wait_tensorcnt((short)0);  // B[cur] DMA done
        __syncthreads();   // A visible; B[cur] visible to all waves
        if (kk + 32 < DIM) {  // kick off next B tiles into the other buffer
            if (wv == 0)
                tdm_load_2d(lds_off(&Bh[cur ^ 1][0][0]),
                            wh + (size_t)col0 * DIM + kk + 32, 64, 32, DIM, DIM, DIM);
            else if (wv == 1)
                tdm_load_2d(lds_off(&Bl[cur ^ 1][0][0]),
                            wl + (size_t)col0 * DIM + kk + 32, 64, 32, DIM, DIM, DIM);
        }
        const int bb = cur;
#else
        {   // fallback: cooperative bf16 copy, 16B per thread per tile
            int r = tid >> 2, part = (tid & 3) << 3;
            *(uint4*)&Bh[0][r][part] = *(const uint4*)(wh + (size_t)(col0 + r) * DIM + kk + part);
            *(uint4*)&Bl[0][r][part] = *(const uint4*)(wl + (size_t)(col0 + r) * DIM + kk + part);
        }
        __syncthreads();
        const int bb = 0;
#endif
        v16bf aH[2], aL[2], bH[2], bL[2];
        for (int t = 0; t < 2; ++t) {
            int ar = wm * 32 + t * 16 + lr;
            aH[t] = load_frag(&Ah[ar][0], base);
            aL[t] = load_frag(&Al[ar][0], base);
            int br = wn * 32 + t * 16 + lr;
            bH[t] = load_frag(&Bh[bb][br][0], base);
            bL[t] = load_frag(&Bl[bb][br][0], base);
        }
        // bf16x3: hi*hi + hi*lo + lo*hi (~fp32 accuracy, 3 WMMAs/tile)
        for (int ti = 0; ti < 2; ++ti)
            for (int tj = 0; tj < 2; ++tj) {
                acc[ti][tj] = __builtin_amdgcn_wmma_f32_16x16x32_bf16(
                    false, aH[ti], false, bH[tj], (short)0, acc[ti][tj], false, false);
                acc[ti][tj] = __builtin_amdgcn_wmma_f32_16x16x32_bf16(
                    false, aH[ti], false, bL[tj], (short)0, acc[ti][tj], false, false);
                acc[ti][tj] = __builtin_amdgcn_wmma_f32_16x16x32_bf16(
                    false, aL[ti], false, bH[tj], (short)0, acc[ti][tj], false, false);
            }
        cur ^= 1;
    }

    // epilogue: C/D layout: element r -> M = r + 8*(lane>=16), N = lane%16
    for (int ti = 0; ti < 2; ++ti)
        for (int tj = 0; tj < 2; ++tj) {
            int colT = col0 + wn * 32 + tj * 16 + lr;
            int rowT = row0 + wm * 32 + ti * 16 + ((lane >> 4) << 3);
            float badd = mode ? bias[colT] : 0.0f;
            for (int r = 0; r < 8; ++r)
                C[(size_t)(rowT + r) * DIM + colT] = acc[ti][tj][r] + badd;
        }
}

// split fp32 weights into hi/lo bf16 once (weights are small and heavily reused)
__global__ void wsplit_kernel(const float* __restrict__ in, __bf16* __restrict__ hi,
                              __bf16* __restrict__ lo, int n) {
    int i = blockIdx.x * 256 + threadIdx.x;
    if (i < n) {
        float v = in[i];
        __bf16 h = f2bf(v);
        hi[i] = h;
        lo[i] = f2bf(v - bf2f(h));
    }
}

// invnorm2(b,c) = 1 / max(sum_n w(b,n,c)^2, 1e-24)
__global__ void colnorm_kernel(const float* __restrict__ w, float* __restrict__ invn2) {
    int blk = blockIdx.x;
    int b = blk / 3;
    int c = (blk % 3) * 256 + threadIdx.x;
    const float* p = w + (size_t)b * NTOK * DIM + c;
    float s = 0.f;
    for (int n = 0; n < NTOK; ++n) { float v = p[(size_t)n * DIM]; s += v * v; }
    invn2[b * DIM + c] = 1.0f / fmaxf(s, 1e-24f);
}

// per token: s_h = temp_h * sum_d w^2*invnorm2 ; softmax over heads -> Pi(b,h,n)
__global__ void pi_kernel(const float* __restrict__ w, const float* __restrict__ invn2,
                          const float* __restrict__ temp, float* __restrict__ Pi) {
    __shared__ float sh[NH];
    int blk = blockIdx.x;
    int b  = blk >> 12;
    int nt = blk & (NTOK - 1);
    int tid = threadIdx.x;
    if (tid < NH) sh[tid] = 0.f;
    __syncthreads();
    const float* row = w + (size_t)blk * DIM;
    const float* inv = invn2 + b * DIM;
    for (int c = tid; c < DIM; c += 256) {
        float v = row[c];
        atomicAdd(&sh[c >> 6], v * v * inv[c]);   // ds_add_f32
    }
    __syncthreads();
    if (tid == 0) {
        float s[NH], m = -1e30f;
        for (int h = 0; h < NH; ++h) { s[h] = temp[h] * sh[h]; m = fmaxf(m, s[h]); }
        float sum = 0.f;
        for (int h = 0; h < NH; ++h) { s[h] = __expf(s[h] - m); sum += s[h]; }
        float rs = 1.0f / sum;
        for (int h = 0; h < NH; ++h)
            Pi[(size_t)(b * NH + h) * NTOK + nt] = s[h] * rs;
    }
}

__global__ void sumpi_kernel(const float* __restrict__ Pi, float* __restrict__ sumPi) {
    __shared__ float red[256];
    int r = blockIdx.x;
    float s = 0.f;
    for (int n = threadIdx.x; n < NTOK; n += 256) s += Pi[(size_t)r * NTOK + n];
    red[threadIdx.x] = s;
    __syncthreads();
    for (int st = 128; st > 0; st >>= 1) {
        if (threadIdx.x < st) red[threadIdx.x] += red[threadIdx.x + st];
        __syncthreads();
    }
    if (threadIdx.x == 0) sumPi[r] = red[0];
}

// attn(b,c) = 1 / (1 + (sum_n Pi*w^2) / (sumPi + 1e-8))
__global__ void attn_kernel(const float* __restrict__ w, const float* __restrict__ Pi,
                            const float* __restrict__ sumPi, float* __restrict__ attn) {
    int blk = blockIdx.x;
    int b = blk / 3;
    int c = (blk % 3) * 256 + threadIdx.x;
    int h = c >> 6;
    const float* p  = w + (size_t)b * NTOK * DIM + c;
    const float* pi = Pi + (size_t)(b * NH + h) * NTOK;
    float s = 0.f;
    for (int n = 0; n < NTOK; ++n) { float v = p[(size_t)n * DIM]; s += pi[n] * v * v; }
    float dots = s / (sumPi[b * NH + h] + 1e-8f);
    attn[b * DIM + c] = 1.0f / (1.0f + dots);
}

extern "C" void kernel_launch(void* const* d_in, const int* in_sizes, int n_in,
                              void* d_out, int out_size, void* d_ws, size_t ws_size,
                              hipStream_t stream) {
    (void)in_sizes; (void)n_in; (void)out_size; (void)ws_size;
    const float* x     = (const float*)d_in[0];
    const float* qkv_w = (const float*)d_in[1];
    const float* temp  = (const float*)d_in[2];
    const float* out_w = (const float*)d_in[3];
    const float* out_b = (const float*)d_in[4];
    float* out = (float*)d_out;

    float* ws    = (float*)d_ws;
    float* w     = ws;                                    // 50,331,648 f
    float* Pi    = w     + (size_t)MTOT * DIM;            //    786,432 f
    float* invn2 = Pi    + (size_t)BSZ * NH * NTOK;       //     12,288 f
    float* sumPi = invn2 + BSZ * DIM;                     //        192 f
    float* attn  = sumPi + BSZ * NH;                      //     12,288 f
    __bf16* qh = (__bf16*)(attn + BSZ * DIM);             //  589,824 bf16 each
    __bf16* ql = qh + (size_t)DIM * DIM;
    __bf16* oh = ql + (size_t)DIM * DIM;
    __bf16* ol = oh + (size_t)DIM * DIM;

    const int NW = DIM * DIM;
    dim3 gg(DIM / 64, MTOT / 128);
    // 0) one-time weight splits (small, reused by every block -> TDM-fed B tiles)
    wsplit_kernel<<<(NW + 255) / 256, 256, 0, stream>>>(qkv_w, qh, ql, NW);
    wsplit_kernel<<<(NW + 255) / 256, 256, 0, stream>>>(out_w, oh, ol, NW);
    // 1) w = x @ qkv_w^T  (bf16x3 WMMA, TDM-fed B)
    gemm_bf16x3_kernel<<<gg, 256, 0, stream>>>(x, qh, ql, w, 0, nullptr, nullptr, nullptr);
    // 2) column L2 norms over tokens
    colnorm_kernel<<<BSZ * 3, 256, 0, stream>>>(w, invn2);
    // 3) per-token head-softmax -> Pi
    pi_kernel<<<MTOT, 256, 0, stream>>>(w, invn2, temp, Pi);
    // 4) sum of Pi over tokens
    sumpi_kernel<<<BSZ * NH, 256, 0, stream>>>(Pi, sumPi);
    // 5) dots + attn = 1/(1+dots)
    attn_kernel<<<BSZ * 3, 256, 0, stream>>>(w, Pi, sumPi, attn);
    // 6) out = (-(w*Pi)*attn) @ out_w^T + out_b, elementwise fused into A-load
    gemm_bf16x3_kernel<<<gg, 256, 0, stream>>>(w, oh, ol, out, 1, Pi, attn, out_b);
}
